// SynTeacher_63290638074042
// MI455X (gfx1250) — matrix operand order repaired and checked by reading, not compiled, not run on hardware
//
#include <hip/hip_runtime.h>

#define INDIM 512
#define HID   128
#define XS_FLOATS (128 * 36)   // one padded x staging buffer (rows x 36 f32)

typedef __attribute__((ext_vector_type(16))) __bf16 v16bf;
typedef __attribute__((ext_vector_type(8)))  float  v8f;
typedef __attribute__((ext_vector_type(4)))  unsigned int u32x4;
typedef __attribute__((ext_vector_type(8)))  int  i32x8;
typedef __attribute__((ext_vector_type(4)))  int  i32x4;

#if defined(__has_builtin)
#if __has_builtin(__builtin_amdgcn_tensor_load_to_lds) && \
    __has_builtin(__builtin_amdgcn_s_wait_tensorcnt)
#define USE_TDM 1
#endif
#endif

// ---------------------------------------------------------------------------
// WMMA helpers: D = A(16x32 bf16) * B(32x16 bf16) + C(16x16 f32)
// 3xBF16 split: a*b ~= ah*bh + ah*bl + al*bh  (fp32-grade accuracy)
// ---------------------------------------------------------------------------
static __device__ __forceinline__ v8f wmma_bf16(v16bf a, v16bf b, v8f c) {
  return __builtin_amdgcn_wmma_f32_16x16x32_bf16(false, a, false, b,
                                                 (short)0, c, false, false);
}

static __device__ __forceinline__ v8f wmma3(v16bf ah, v16bf al,
                                            v16bf bh, v16bf bl, v8f c) {
  c = wmma_bf16(ah, bh, c);
  c = wmma_bf16(ah, bl, c);
  c = wmma_bf16(al, bh, c);
  return c;
}

// A-fragment (16-bit A 16x32 layout, ISA 7.12.2):
//  lanes 0-15 : elements 0..7 -> K=0..7,  8..15 -> K=16..23
//  lanes 16-31: elements 0..7 -> K=8..15, 8..15 -> K=24..31
static __device__ __forceinline__ void load_a_frag(const float* p,
                                                   v16bf& ah, v16bf& al) {
#pragma unroll
  for (int e = 0; e < 8; ++e) {
    float v0 = p[e];
    float v1 = p[16 + e];
    __bf16 h0 = (__bf16)v0;
    __bf16 h1 = (__bf16)v1;
    ah[e]     = h0;  al[e]     = (__bf16)(v0 - (float)h0);
    ah[8 + e] = h1;  al[8 + e] = (__bf16)(v1 - (float)h1);
  }
}

static __device__ __forceinline__ void zero_acc(v8f* acc) {
  v8f z = {0.f, 0.f, 0.f, 0.f, 0.f, 0.f, 0.f, 0.f};
#pragma unroll
  for (int i = 0; i < 8; ++i) acc[i] = z;
}

static __device__ __forceinline__ void gemm_lds(const float* hsw, int kchunks,
                                                const __bf16* wh, const __bf16* wl,
                                                v8f* acc, int nn, int kh, int lane) {
  for (int kc = 0; kc < kchunks; ++kc) {
    v16bf ah, al;
    load_a_frag(hsw + nn * 264 + kc * 32 + kh * 8, ah, al);
#pragma unroll
    for (int nt = 0; nt < 8; ++nt) {
      v16bf bh = ((const v16bf*)wh)[(kc * 8 + nt) * 32 + lane];
      v16bf bl = ((const v16bf*)wl)[(kc * 8 + nt) * 32 + lane];
      acc[nt] = wmma3(ah, al, bh, bl, acc[nt]);
    }
  }
}

// C/D 16x16 f32 layout: lane = (m>=8)*16 + n%16 ; vgpr r -> m = r + 8*(lane>=16)
static __device__ __forceinline__ void tile_store_lds(float* dst, const v8f* acc,
                                                      const float* bias, bool relu,
                                                      int nn, int kh) {
#pragma unroll
  for (int nt = 0; nt < 8; ++nt) {
    float bv = bias[nt * 16 + nn];
#pragma unroll
    for (int r = 0; r < 8; ++r) {
      float v = acc[nt][r] + bv;
      if (relu) v = fmaxf(v, 0.0f);
      dst[(r + 8 * kh) * 264 + nt * 16 + nn] = v;
    }
  }
}

#ifdef USE_TDM
// ---------------------------------------------------------------------------
// Tensor Data Mover issue: 2D tile (tile_dim0=32 cols x tile_dim1=128 rows),
// f32 elements, tensor_dim0_stride=512, LDS padding 4 DWORDs per 32 DWORDs
// (-> 36-float padded rows). OOB rows (y >= rows_rem) read as zero.
// D# layout per CDNA5 ISA 8.3/8.4. 6-arg builtin (clang-23 lane).
// ---------------------------------------------------------------------------
static __device__ __forceinline__ void tdm_issue(unsigned long long gaddr,
                                                 unsigned lds_off, int rows_rem) {
  u32x4 g0;
  g0[0] = 1u;                                   // count=1 (valid), load, user
  g0[1] = lds_off;                              // lds_addr [63:32]
  g0[2] = (unsigned)(gaddr & 0xFFFFFFFFu);      // global_addr [95:64]
  g0[3] = (unsigned)((gaddr >> 32) & 0x1FFFFFFu) | (2u << 30); // [120:96]+type=2
  i32x8 g1;
  g1[0] = (2 << 16)        // data_size = 4 bytes
        | (1 << 20)        // pad_enable
        | (4 << 22)        // pad_interval: 2^(4+1)=32 DWORDs
        | (3 << 25);       // pad_amount: 3+1 = 4 DWORDs
  g1[1] = (int)(512u << 16);                    // tensor_dim0[15:0] @ [63:48]
  g1[2] = (int)(((unsigned)rows_rem & 0xFFFFu) << 16); // dim0 hi=0, dim1 lo
  g1[3] = (int)((((unsigned)rows_rem >> 16) & 0xFFFFu) | (32u << 16)); // dim1 hi, tile_dim0=32
  g1[4] = 128;                                  // tile_dim1=128, tile_dim2=0
  g1[5] = 512;                                  // tensor_dim0_stride[31:0]
  g1[6] = 0;                                    // stride hi, dim1_stride lo
  g1[7] = 0;
  i32x4 gz4 = {0, 0, 0, 0};                     // 2D tensor: groups 2/3 unused
  i32x8 gz8 = {0, 0, 0, 0, 0, 0, 0, 0};
  __builtin_amdgcn_tensor_load_to_lds(g0, g1, gz4, gz4, gz8, 0);
}
#endif

// ---------------------------------------------------------------------------
// Prep kernels
// ---------------------------------------------------------------------------
// Pack W[K][128] (row-major f32) into fragment-ordered bf16 hi/lo:
// B 32x16 tile layout: lane = ((kk>>3)&1)*16 + n%16, element e = kk%8 + 8*(kk>=16).
__global__ void k_pack(const float* __restrict__ W, int K,
                       __bf16* __restrict__ hi, __bf16* __restrict__ lo) {
  int idx = blockIdx.x * blockDim.x + threadIdx.x;
  if (idx >= K * HID) return;
  int k = idx / HID, n = idx % HID;
  int kc = k >> 5, kk = k & 31, nt = n >> 4, nn = n & 15;
  int lane = (((kk >> 3) & 1) << 4) | nn;
  int e    = (kk & 7) | ((kk >> 4) << 3);
  int slot = (((kc * 8 + nt) * 32 + lane) << 4) + e;
  float v  = W[idx];
  __bf16 h = (__bf16)v;
  hi[slot] = h;
  lo[slot] = (__bf16)(v - (float)h);
}

__global__ void k_colsum(const float* __restrict__ Wg, float* __restrict__ vcol) {
  int j = threadIdx.x;  // 128 threads
  float s = 0.0f;
  for (int i = 0; i < INDIM; ++i) s += Wg[i * HID + j];
  vcol[j] = s;
}

__global__ void k_init(float* __restrict__ deg, float* __restrict__ tac, int N) {
  int i = blockIdx.x * blockDim.x + threadIdx.x;
  if (i < N) { deg[i] = 1.0f; tac[i] = 0.0f; }  // self-loop contributes 1
}

__global__ void k_degree(const int* __restrict__ dst, int E, float* __restrict__ deg) {
  int i = blockIdx.x * blockDim.x + threadIdx.x;
  if (i < E) atomicAdd(&deg[dst[i]], 1.0f);  // integer-valued: exact in f32
}

__global__ void k_norm(const float* __restrict__ deg, float* __restrict__ nrm, int N) {
  int i = blockIdx.x * blockDim.x + threadIdx.x;
  if (i < N) nrm[i] = rsqrtf(deg[i]);
}

__global__ void k_scatter(const int* __restrict__ src, const int* __restrict__ dst,
                          int E, const float* __restrict__ nrm, float* __restrict__ tac) {
  int i = blockIdx.x * blockDim.x + threadIdx.x;
  if (i < E) atomicAdd(&tac[dst[i]], nrm[src[i]]);
}

// ---------------------------------------------------------------------------
// Fused main kernel: 8 waves/block, 16 rows/wave (128 rows/block).
// stage1: h1 = relu(x@W1+b1)  (K=512, x staged to LDS by TDM, double-buffered)
// stage2: h1 = relu(h1@W2+b2) ; h2 = s*v + bg (rank-1 GCN) ; h=[h1,h2]
// proj  : ((h@Wp1+bp1)@Wp2+bp2)@Wp3+bp3 -> out
// ---------------------------------------------------------------------------
__global__ __launch_bounds__(256) void k_fused(
    const float* __restrict__ x,
    const float* __restrict__ b1, const float* __restrict__ b2,
    const float* __restrict__ bg,
    const float* __restrict__ bp1, const float* __restrict__ bp2,
    const float* __restrict__ bp3,
    const __bf16* __restrict__ w1h, const __bf16* __restrict__ w1l,
    const __bf16* __restrict__ w2h, const __bf16* __restrict__ w2l,
    const __bf16* __restrict__ wp1h, const __bf16* __restrict__ wp1l,
    const __bf16* __restrict__ wp2h, const __bf16* __restrict__ wp2l,
    const __bf16* __restrict__ wp3h, const __bf16* __restrict__ wp3l,
    const float* __restrict__ nrm, const float* __restrict__ tac,
    const float* __restrict__ vcol,
    float* __restrict__ out, int N) {
  extern __shared__ float smem[];
  float* xs = smem;                      // 2 x (128 rows x 36 f32), double buffer
  float* hs = smem + 2 * XS_FLOATS;      // 8 waves x (16 rows x 264 f32)

  const int tid  = threadIdx.x;
  const int lane = tid & 31;
  const int w    = tid >> 5;
  const int nn   = lane & 15;
  const int kh   = lane >> 4;
  const int rbase0 = blockIdx.x * 128;
  const int rwave  = rbase0 + w * 16;
  float* hsw = hs + w * (16 * 264);

  v8f acc[8];
  zero_acc(acc);

  // ---- stage 1: x @ W1, K = 512 in 16 chunks of 32 ----
#ifdef USE_TDM
  const unsigned long long xbase =
      (unsigned long long)(const unsigned char*)x +
      (unsigned long long)rbase0 * (INDIM * 4);
  const int rows_rem = N - rbase0;       // TDM zero-fills OOB rows: no clamping
  if (w == 0) tdm_issue(xbase, 0u, rows_rem);
  for (int kc = 0; kc < 16; ++kc) {
    if (w == 0) __builtin_amdgcn_s_wait_tensorcnt((short)0);  // chunk kc landed
    __syncthreads();                     // publish buffer kc to all waves
    if (w == 0 && kc < 15)               // overlap transfer kc+1 with compute kc
      tdm_issue(xbase + (unsigned long long)(kc + 1) * 128,
                (unsigned)(((kc + 1) & 1) * (XS_FLOATS * 4)), rows_rem);
    const float* xbuf = xs + (kc & 1) * XS_FLOATS;
    v16bf ah, al;
    load_a_frag(xbuf + (w * 16 + nn) * 36 + kh * 8, ah, al);
#pragma unroll
    for (int nt = 0; nt < 8; ++nt) {
      v16bf bh = ((const v16bf*)w1h)[(kc * 8 + nt) * 32 + lane];
      v16bf bl = ((const v16bf*)w1l)[(kc * 8 + nt) * 32 + lane];
      acc[nt] = wmma3(ah, al, bh, bl, acc[nt]);
    }
  }
#else
  for (int kc = 0; kc < 16; ++kc) {
    __syncthreads();
#pragma unroll
    for (int it = 0; it < 4; ++it) {     // cooperative fallback staging
      int idx = it * 256 + tid;
      int row = idx >> 3, q = idx & 7;
      int grow = rbase0 + row; if (grow >= N) grow = N - 1;
      const float4 vv = *(const float4*)(x + (size_t)grow * INDIM + kc * 32 + q * 4);
      *(float4*)(xs + row * 36 + q * 4) = vv;
      if (kc < 15)
        __builtin_prefetch(x + (size_t)grow * INDIM + (kc + 1) * 32 + q * 4, 0, 1);
    }
    __syncthreads();
    v16bf ah, al;
    load_a_frag(xs + (w * 16 + nn) * 36 + kh * 8, ah, al);
#pragma unroll
    for (int nt = 0; nt < 8; ++nt) {
      v16bf bh = ((const v16bf*)w1h)[(kc * 8 + nt) * 32 + lane];
      v16bf bl = ((const v16bf*)w1l)[(kc * 8 + nt) * 32 + lane];
      acc[nt] = wmma3(ah, al, bh, bl, acc[nt]);
    }
  }
#endif
  tile_store_lds(hsw, acc, b1, /*relu=*/true, nn, kh);

  // ---- stage 2: h1 @ W2 (K=128) ----
  zero_acc(acc);
  gemm_lds(hsw, 4, w2h, w2l, acc, nn, kh, lane);
  tile_store_lds(hsw, acc, b2, /*relu=*/true, nn, kh);

  // ---- GCN branch (rank-1): h2 = s * vcol + bg -> hs cols 128..255 ----
  {
    int m = nn;
    int grow = rwave + m; if (grow >= N) grow = N - 1;
    float nv = nrm[grow];
    float sm = nv * (tac[grow] + nv);
    for (int j = kh * 64; j < kh * 64 + 64; ++j)
      hsw[m * 264 + 128 + j] = sm * vcol[j] + bg[j];
  }

  // ---- projector: K=256, then K=128, K=128 (linear, no relu) ----
  zero_acc(acc);
  gemm_lds(hsw, 8, wp1h, wp1l, acc, nn, kh, lane);
  tile_store_lds(hsw, acc, bp1, /*relu=*/false, nn, kh);

  zero_acc(acc);
  gemm_lds(hsw, 4, wp2h, wp2l, acc, nn, kh, lane);
  tile_store_lds(hsw, acc, bp2, /*relu=*/false, nn, kh);

  zero_acc(acc);
  gemm_lds(hsw, 4, wp3h, wp3l, acc, nn, kh, lane);

  // ---- final store (+bp3), masked for ragged tail ----
#pragma unroll
  for (int nt = 0; nt < 8; ++nt) {
    float bv = bp3[nt * 16 + nn];
#pragma unroll
    for (int r = 0; r < 8; ++r) {
      int grow = rwave + r + 8 * kh;
      if (grow < N)
        out[(size_t)grow * HID + nt * 16 + nn] = acc[nt][r] + bv;
    }
  }
}

// ---------------------------------------------------------------------------
extern "C" void kernel_launch(void* const* d_in, const int* in_sizes, int n_in,
                              void* d_out, int out_size, void* d_ws, size_t ws_size,
                              hipStream_t stream) {
  const float* x   = (const float*)d_in[0];
  // d_in[1] = x_ones: unused (algebraically replaced by colsum(Wg))
  const int*   ei  = (const int*)d_in[2];
  const float* W1  = (const float*)d_in[3];  const float* b1  = (const float*)d_in[4];
  const float* W2  = (const float*)d_in[5];  const float* b2  = (const float*)d_in[6];
  const float* Wg  = (const float*)d_in[7];  const float* bg  = (const float*)d_in[8];
  const float* Wp1 = (const float*)d_in[9];  const float* bp1 = (const float*)d_in[10];
  const float* Wp2 = (const float*)d_in[11]; const float* bp2 = (const float*)d_in[12];
  const float* Wp3 = (const float*)d_in[13]; const float* bp3 = (const float*)d_in[14];
  float* out = (float*)d_out;

  const int N = in_sizes[0] / INDIM;
  const int E = in_sizes[2] / 2;
  const int* src = ei;
  const int* dst = ei + E;

  // workspace partition (256B aligned regions)
  char* ws = (char*)d_ws;
  auto alloc = [&](size_t bytes) {
    char* p = ws; ws += (bytes + 255) & ~(size_t)255; return p;
  };
  float* deg  = (float*)alloc((size_t)N * 4);
  float* nrm  = (float*)alloc((size_t)N * 4);
  float* tac  = (float*)alloc((size_t)N * 4);
  float* vcol = (float*)alloc(HID * 4);
  __bf16* w1h  = (__bf16*)alloc((size_t)INDIM * HID * 2);
  __bf16* w1l  = (__bf16*)alloc((size_t)INDIM * HID * 2);
  __bf16* w2h  = (__bf16*)alloc((size_t)HID * HID * 2);
  __bf16* w2l  = (__bf16*)alloc((size_t)HID * HID * 2);
  __bf16* wp1h = (__bf16*)alloc((size_t)2 * HID * HID * 2);
  __bf16* wp1l = (__bf16*)alloc((size_t)2 * HID * HID * 2);
  __bf16* wp2h = (__bf16*)alloc((size_t)HID * HID * 2);
  __bf16* wp2l = (__bf16*)alloc((size_t)HID * HID * 2);
  __bf16* wp3h = (__bf16*)alloc((size_t)HID * HID * 2);
  __bf16* wp3l = (__bf16*)alloc((size_t)HID * HID * 2);

  // --- weight packing (hi/lo bf16, fragment order) ---
  k_pack<<<(INDIM * HID + 255) / 256, 256, 0, stream>>>(W1, INDIM, w1h, w1l);
  k_pack<<<(HID * HID + 255) / 256, 256, 0, stream>>>(W2, HID, w2h, w2l);
  k_pack<<<(2 * HID * HID + 255) / 256, 256, 0, stream>>>(Wp1, 2 * HID, wp1h, wp1l);
  k_pack<<<(HID * HID + 255) / 256, 256, 0, stream>>>(Wp2, HID, wp2h, wp2l);
  k_pack<<<(HID * HID + 255) / 256, 256, 0, stream>>>(Wp3, HID, wp3h, wp3l);
  k_colsum<<<1, HID, 0, stream>>>(Wg, vcol);

  // --- GCN scalar pipeline: deg -> norm -> t ---
  k_init<<<(N + 255) / 256, 256, 0, stream>>>(deg, tac, N);
  k_degree<<<(E + 255) / 256, 256, 0, stream>>>(dst, E, deg);
  k_norm<<<(N + 255) / 256, 256, 0, stream>>>(deg, nrm, N);
  k_scatter<<<(E + 255) / 256, 256, 0, stream>>>(src, dst, E, nrm, tac);

  // --- fused WMMA pipeline ---
  const int    blocks = (N + 127) / 128;
  const size_t shmem  = (size_t)(2 * XS_FLOATS + 8 * 16 * 264) * sizeof(float);
  k_fused<<<blocks, 256, shmem, stream>>>(
      x, b1, b2, bg, bp1, bp2, bp3,
      w1h, w1l, w2h, w2l, wp1h, wp1l, wp2h, wp2l, wp3h, wp3l,
      nrm, tac, vcol, out, N);
}
